// EdgeModel_17910013624368
// MI455X (gfx1250) — matrix operand order repaired and checked by reading, not compiled, not run on hardware
//
#include <hip/hip_runtime.h>

typedef __attribute__((ext_vector_type(16))) _Float16 v16h;
typedef __attribute__((ext_vector_type(8)))  _Float16 v8h;
typedef __attribute__((ext_vector_type(8)))  float    v8f;
typedef __attribute__((ext_vector_type(4)))  float    v4f;

#define DH 128           // hidden dim
#define K0 384           // layer-0 K (3*DH)
#define TILE_M 128       // edges per workgroup
#define W0S 400          // LDS row stride (halves) for W0: 384 + 16 pad -> 8-bank lane stride
#define W1S 144          // LDS row stride (halves) for W1/W2: 128 + 16 pad
#define HS  136          // LDS row stride (halves) for h: 128 + 8 pad

__device__ __forceinline__ float silu_f(float x) {
    return x / (1.0f + __expf(-x));
}

// Convert W [K x N] fp32 row-major -> WT [N x K] fp16 row-major (B-fragment friendly).
__global__ void cvt_transpose_kernel(const float* __restrict__ W,
                                     _Float16* __restrict__ WT, int K, int N) {
    int idx = blockIdx.x * blockDim.x + threadIdx.x;
    if (idx >= K * N) return;
    int n = idx / K;
    int k = idx - n * K;
    WT[idx] = (_Float16)W[k * N + n];
}

__global__ __launch_bounds__(256)
void edge_mlp_wmma(const float* __restrict__ src, const float* __restrict__ dst,
                   const float* __restrict__ ea,
                   const _Float16* __restrict__ W0T,  // [128 x 384] fp16
                   const _Float16* __restrict__ W1T,  // [128 x 128] fp16
                   const _Float16* __restrict__ W2T,  // [128 x 128] fp16
                   const float* __restrict__ b0, const float* __restrict__ b1,
                   const float* __restrict__ b2,
                   float* __restrict__ out, int E) {
    // ~206 KB of LDS: one workgroup per WGP (CDNA5: 320 KB/WG).
    __shared__ __align__(32) _Float16 sW0[DH * W0S];   // 102400 B
    __shared__ __align__(32) _Float16 sW1[DH * W1S];   //  36864 B
    __shared__ __align__(32) _Float16 sW2[DH * W1S];   //  36864 B
    __shared__ __align__(32) _Float16 hBuf[TILE_M * HS]; // 34816 B

    const int tid  = threadIdx.x;
    const int wave = tid >> 5;           // 0..7
    const int lane = tid & 31;
    const int lrow = lane & 15;          // M (A) or N (B/C) index inside a 16-tile
    const int lhi  = lane >> 4;          // half-wave selector

    // 2 M-tiles x 4 N-tiles per wave: waves arranged 4 (M-groups) x 2 (N-groups)
    const int mBase = (wave & 3) * 32;   // rows [mBase, mBase+32)
    const int nBase = (wave >> 2) * 64;  // cols [nBase, nBase+64)
    const long long rowBase = (long long)blockIdx.x * TILE_M;

    // ---- stage fp16 transposed weights into LDS (once per block) ----
    for (int c = tid; c < DH * (K0 / 8); c += 256) {     // W0: 128 rows x 48 chunks
        int r = c / (K0 / 8), cc = c - r * (K0 / 8);
        *(v8h*)(sW0 + r * W0S + cc * 8) = *(const v8h*)(W0T + r * K0 + cc * 8);
    }
    for (int c = tid; c < DH * (DH / 8); c += 256) {     // W1/W2: 128 rows x 16 chunks
        int r = c >> 4, cc = c & 15;
        *(v8h*)(sW1 + r * W1S + cc * 8) = *(const v8h*)(W1T + r * DH + cc * 8);
        *(v8h*)(sW2 + r * W1S + cc * 8) = *(const v8h*)(W2T + r * DH + cc * 8);
    }

    // Clamped global rows for the two A fragments (garbage rows replicate row E-1).
    long long arow0 = rowBase + mBase + lrow;        if (arow0 >= E) arow0 = E - 1;
    long long arow1 = rowBase + mBase + 16 + lrow;   if (arow1 >= E) arow1 = E - 1;
    const float* aseg[2][3] = {
        { ea + arow0 * DH, src + arow0 * DH, dst + arow0 * DH },
        { ea + arow1 * DH, src + arow1 * DH, dst + arow1 * DH },
    };

    __syncthreads();   // weights staged

    v8f acc[2][4];

    // ---------------- Layer 0: x(E x 384) @ W0 ----------------
    #pragma unroll
    for (int mi = 0; mi < 2; ++mi)
        #pragma unroll
        for (int ni = 0; ni < 4; ++ni) acc[mi][ni] = (v8f){};

    #pragma unroll
    for (int kt = 0; kt < 12; ++kt) {
        v16h a[2];
        #pragma unroll
        for (int mi = 0; mi < 2; ++mi) {
            const float* p = aseg[mi][kt >> 2] + (kt & 3) * 32 + lhi * 16;
            v4f x0 = *(const v4f*)(p);
            v4f x1 = *(const v4f*)(p + 4);
            v4f x2 = *(const v4f*)(p + 8);
            v4f x3 = *(const v4f*)(p + 12);
            #pragma unroll
            for (int i = 0; i < 4; ++i) {
                a[mi][i]      = (_Float16)x0[i];
                a[mi][4 + i]  = (_Float16)x1[i];
                a[mi][8 + i]  = (_Float16)x2[i];
                a[mi][12 + i] = (_Float16)x3[i];
            }
        }
        v16h b[4];
        #pragma unroll
        for (int ni = 0; ni < 4; ++ni)
            b[ni] = *(const v16h*)(sW0 + (nBase + ni * 16 + lrow) * W0S + kt * 32 + lhi * 16);
        #pragma unroll
        for (int mi = 0; mi < 2; ++mi)
            #pragma unroll
            for (int ni = 0; ni < 4; ++ni)
                acc[mi][ni] = __builtin_amdgcn_wmma_f32_16x16x32_f16(
                    false, a[mi], false, b[ni], (short)0, acc[mi][ni], false, false);
    }

    // bias + silu -> fp16 -> hBuf
    #pragma unroll
    for (int ni = 0; ni < 4; ++ni) {
        float bv = b0[nBase + ni * 16 + lrow];
        #pragma unroll
        for (int mi = 0; mi < 2; ++mi)
            #pragma unroll
            for (int g = 0; g < 8; ++g) {
                int m = mBase + mi * 16 + lhi * 8 + g;
                hBuf[m * HS + nBase + ni * 16 + lrow] = (_Float16)silu_f(acc[mi][ni][g] + bv);
            }
    }
    __syncthreads();

    // ---------------- Layers 1 & 2: h @ W ----------------
    #pragma unroll
    for (int layer = 1; layer <= 2; ++layer) {
        const _Float16* sW = (layer == 1) ? sW1 : sW2;
        const float*    bb = (layer == 1) ? b1  : b2;

        #pragma unroll
        for (int mi = 0; mi < 2; ++mi)
            #pragma unroll
            for (int ni = 0; ni < 4; ++ni) acc[mi][ni] = (v8f){};

        #pragma unroll
        for (int kt = 0; kt < 4; ++kt) {
            v16h a[2];
            #pragma unroll
            for (int mi = 0; mi < 2; ++mi) {
                const _Float16* ap = hBuf + (mBase + mi * 16 + lrow) * HS + kt * 32 + lhi * 16;
                v8h alo = *(const v8h*)(ap);
                v8h ahi = *(const v8h*)(ap + 8);
                #pragma unroll
                for (int i = 0; i < 8; ++i) { a[mi][i] = alo[i]; a[mi][8 + i] = ahi[i]; }
            }
            v16h b[4];
            #pragma unroll
            for (int ni = 0; ni < 4; ++ni)
                b[ni] = *(const v16h*)(sW + (nBase + ni * 16 + lrow) * W1S + kt * 32 + lhi * 16);
            #pragma unroll
            for (int mi = 0; mi < 2; ++mi)
                #pragma unroll
                for (int ni = 0; ni < 4; ++ni)
                    acc[mi][ni] = __builtin_amdgcn_wmma_f32_16x16x32_f16(
                        false, a[mi], false, b[ni], (short)0, acc[mi][ni], false, false);
        }

        if (layer == 1) {
            __syncthreads();  // all hBuf reads done before overwrite
            #pragma unroll
            for (int ni = 0; ni < 4; ++ni) {
                float bv = bb[nBase + ni * 16 + lrow];
                #pragma unroll
                for (int mi = 0; mi < 2; ++mi)
                    #pragma unroll
                    for (int g = 0; g < 8; ++g) {
                        int m = mBase + mi * 16 + lhi * 8 + g;
                        hBuf[m * HS + nBase + ni * 16 + lrow] =
                            (_Float16)silu_f(acc[mi][ni][g] + bv);
                    }
            }
            __syncthreads();
        } else {
            // final: bias, predicated global store (divergence only after all WMMAs)
            #pragma unroll
            for (int ni = 0; ni < 4; ++ni) {
                float bv = bb[nBase + ni * 16 + lrow];
                #pragma unroll
                for (int mi = 0; mi < 2; ++mi)
                    #pragma unroll
                    for (int g = 0; g < 8; ++g) {
                        long long row = rowBase + mBase + mi * 16 + lhi * 8 + g;
                        if (row < E)
                            out[row * DH + nBase + ni * 16 + lrow] = acc[mi][ni][g] + bv;
                    }
            }
        }
    }
}

extern "C" void kernel_launch(void* const* d_in, const int* in_sizes, int n_in,
                              void* d_out, int out_size, void* d_ws, size_t ws_size,
                              hipStream_t stream) {
    const float* src = (const float*)d_in[0];
    const float* dst = (const float*)d_in[1];
    const float* ea  = (const float*)d_in[2];
    const float* W0  = (const float*)d_in[3];
    const float* b0  = (const float*)d_in[4];
    const float* W1  = (const float*)d_in[5];
    const float* b1  = (const float*)d_in[6];
    const float* W2  = (const float*)d_in[7];
    const float* b2  = (const float*)d_in[8];
    float* out = (float*)d_out;

    const int E = in_sizes[0] / DH;   // 500000

    // fp16 transposed weights in workspace: 96KB + 32KB + 32KB
    _Float16* W0T = (_Float16*)d_ws;
    _Float16* W1T = W0T + K0 * DH;
    _Float16* W2T = W1T + DH * DH;

    cvt_transpose_kernel<<<(K0 * DH + 255) / 256, 256, 0, stream>>>(W0, W0T, K0, DH);
    cvt_transpose_kernel<<<(DH * DH + 255) / 256, 256, 0, stream>>>(W1, W1T, DH, DH);
    cvt_transpose_kernel<<<(DH * DH + 255) / 256, 256, 0, stream>>>(W2, W2T, DH, DH);

    const int nblocks = (E + TILE_M - 1) / TILE_M;
    edge_mlp_wmma<<<nblocks, 256, 0, stream>>>(src, dst, ea, W0T, W1T, W2T,
                                               b0, b1, b2, out, E);
}
